// C2VecLayer_4337916969641
// MI455X (gfx1250) — compile-verified, matrix operand order checked by reading, not run on hardware
//
#include <hip/hip_runtime.h>
#include <hip/hip_bf16.h>

// ---------------------------------------------------------------------------
// Fused bidirectional-LSTM scan for MI455X (gfx1250, wave32, WMMA).
//   z = sigmoid([x_t | h] @ [W;R] + b);  c = f*c + i*g;  h = o*sigmoid(c)
// Grid: 4 (input x direction) combos * 16 batch tiles of 16 rows = 64 blocks.
// Block: 256 threads = 8 waves; 13 N-tiles (4H=200 -> 208) round-robined.
// K = F+H = 100 -> padded 128 -> 4 bf16 WMMA k-steps of 32.
// Sigmoid via v_exp_f32 + v_rcp_f32 (no IEEE division on the scan's
// serial critical path).
// ---------------------------------------------------------------------------

typedef __attribute__((ext_vector_type(16))) __bf16 v16bf;
typedef __attribute__((ext_vector_type(8)))  float  v8f;

#define B_    256
#define T_    766
#define F_    50
#define H_    50
#define G4    200          // 4*H
#define NT    13           // N tiles of 16 (200 -> 208)
#define NPAD  208
#define KS    4            // k-steps of 32 (K = 100 -> 128)

#define BFRAG_BYTES (KS*NT*32*16*2)   // 53248  pre-swizzled B fragments (bf16)
#define AFRAG_BYTES (KS*32*16*2)      // 4096   [x|h] A fragments (bf16)
#define ZBUF_BYTES  (16*NPAD*4)       // 13312  gate buffer (f32)
#define CBUF_BYTES  (16*H_*4)         // 3200   cell state (f32)
#define BIAS_BYTES  (NPAD*4)          // 832
#define SMEM_BYTES  (BFRAG_BYTES+AFRAG_BYTES+ZBUF_BYTES+CBUF_BYTES+BIAS_BYTES)

__device__ __forceinline__ __bf16 f2bf(float f) {
  union { float f; unsigned u; } in; in.f = f;
  unsigned u = in.u;
  unsigned r = u + 0x7FFFu + ((u >> 16) & 1u);     // round-to-nearest-even
  union { unsigned short s; __bf16 b; } out; out.s = (unsigned short)(r >> 16);
  return out.b;
}

// Fast sigmoid: v_exp_f32 computes 2^x, so sigmoid(x) = rcp(1 + 2^(-x*log2e)).
// v_rcp_f32 is ~1 ulp — far below bf16 WMMA quantization already in the loop.
#define LOG2E 1.44269504088896340736f
__device__ __forceinline__ float sigmoidf_(float x) {
  float e = __builtin_amdgcn_exp2f(x * -LOG2E);
  return __builtin_amdgcn_rcpf(1.0f + e);
}

// ISA 7.12.2: 16-bit A matrix 16x32 per k-step. For element (row m, depth K):
//   ks = K/32; within step: VGPR quad = (K%32)/16, lane half = (K%16)/8,
//   vgpr-in-quad = ((K%8)/2), packed half = K&1; lane = half*16 + m.
__device__ __forceinline__ void store_afrag(__bf16* Afrag, int m, int K, __bf16 v) {
  int ks    = K >> 5;
  int r     = K & 31;
  int grp   = r >> 4;          // VGPR quad (0..1)
  int r2    = r & 15;
  int lhalf = r2 >> 3;         // lane half
  int r3    = r2 & 7;
  int j     = grp * 4 + (r3 >> 1);   // VGPR index 0..7
  int e     = r3 & 1;                // 16-bit half within VGPR
  int lane  = lhalf * 16 + m;
  Afrag[(ks * 32 + lane) * 16 + j * 2 + e] = v;
}

__global__ void bilstm_scan_kernel(const float* __restrict__ context,
                                   const float* __restrict__ question,
                                   const float* __restrict__ Wf,
                                   const float* __restrict__ Rf,
                                   const float* __restrict__ bfv,
                                   const float* __restrict__ Wb,
                                   const float* __restrict__ Rb,
                                   const float* __restrict__ bbv,
                                   float* __restrict__ out) {
  extern __shared__ unsigned char smem[];
  __bf16* Bfrag = (__bf16*)(smem);
  __bf16* Afrag = (__bf16*)(smem + BFRAG_BYTES);
  float*  zbuf  = (float*)(smem + BFRAG_BYTES + AFRAG_BYTES);
  float*  cbuf  = (float*)(smem + BFRAG_BYTES + AFRAG_BYTES + ZBUF_BYTES);
  float*  biasl = (float*)(smem + BFRAG_BYTES + AFRAG_BYTES + ZBUF_BYTES + CBUF_BYTES);

  const int tid     = threadIdx.x;
  const int lane    = tid & 31;
  const int wv      = tid >> 5;
  const int combo   = blockIdx.x >> 4;   // 0..3
  const int btile   = blockIdx.x & 15;   // 0..15
  const int dir     = combo & 1;         // 0 = fwd scan, 1 = bwd scan
  const int inp     = combo >> 1;        // 0 = context, 1 = question
  const int rowbase = btile * 16;

  const float* x  = inp ? question : context;
  const float* W  = dir ? Wb  : Wf;
  const float* R  = dir ? Rb  : Rf;
  const float* bb = dir ? bbv : bfv;

  // -------- one-time staging: swizzle [W;R] (100x200 -> 128x208) into
  // per-lane WMMA B-fragment order (mirror of A striping with M<->N) -------
  for (int idx = tid; idx < KS * NT * 32 * 16; idx += blockDim.x) {
    int e  = idx & 15;
    int l  = (idx >> 4) & 31;
    int nt = (idx >> 9) % NT;
    int ks = idx / (NT * 512);
    int j  = e >> 1, eo = e & 1;
    int K  = ks * 32 + (j >> 2) * 16 + (l >> 4) * 8 + (j & 3) * 2 + eo;
    int n  = nt * 16 + (l & 15);
    float v = 0.0f;
    if (n < G4) {
      if (K < F_)            v = W[K * G4 + n];
      else if (K < F_ + H_)  v = R[(K - F_) * G4 + n];
    }
    Bfrag[idx] = f2bf(v);
  }
  for (int idx = tid; idx < KS * 32 * 16; idx += blockDim.x) Afrag[idx] = f2bf(0.0f); // h0 = 0, K-pad = 0
  for (int idx = tid; idx < 16 * H_;     idx += blockDim.x) cbuf[idx] = 0.0f;         // c0 = 0
  for (int idx = tid; idx < NPAD;        idx += blockDim.x) biasl[idx] = (idx < G4) ? bb[idx] : 0.0f;
  __syncthreads();

  for (int s = 0; s < T_; ++s) {
    const int t = dir ? (T_ - 1 - s) : s;

    // ---- phase 1: stream x_t (16 rows x 50) into A fragments (K = 0..49) ----
    for (int idx = tid; idx < 16 * F_; idx += blockDim.x) {
      int m = idx / F_, k = idx % F_;
      int rg = rowbase + m;
      float xv = x[((size_t)rg * T_ + t) * F_ + k];
      store_afrag(Afrag, m, k, f2bf(xv));
      if (s + 1 < T_) {                       // hide next step's HBM latency
        int tn = dir ? (t - 1) : (t + 1);
        __builtin_prefetch(&x[((size_t)rg * T_ + tn) * F_ + k], 0, 1);
      }
    }
    __syncthreads();

    // ---- phase 2: z = sigmoid([x|h] @ [W;R] + b) via bf16 WMMA ----
    {
      const v16bf* Ab = (const v16bf*)Afrag;
      v16bf a0 = Ab[0 * 32 + lane];
      v16bf a1 = Ab[1 * 32 + lane];
      v16bf a2 = Ab[2 * 32 + lane];
      v16bf a3 = Ab[3 * 32 + lane];
      const v16bf* Bb = (const v16bf*)Bfrag;
      for (int nt = wv; nt < NT; nt += 8) {
        float bv = biasl[nt * 16 + (lane & 15)];   // C/D: N = lane%16 in every VGPR
        v8f acc;
        for (int r = 0; r < 8; ++r) acc[r] = bv;
        acc = __builtin_amdgcn_wmma_f32_16x16x32_bf16(false, a0, false, Bb[(0 * NT + nt) * 32 + lane], (short)0, acc, false, false);
        acc = __builtin_amdgcn_wmma_f32_16x16x32_bf16(false, a1, false, Bb[(1 * NT + nt) * 32 + lane], (short)0, acc, false, false);
        acc = __builtin_amdgcn_wmma_f32_16x16x32_bf16(false, a2, false, Bb[(2 * NT + nt) * 32 + lane], (short)0, acc, false, false);
        acc = __builtin_amdgcn_wmma_f32_16x16x32_bf16(false, a3, false, Bb[(3 * NT + nt) * 32 + lane], (short)0, acc, false, false);
        int col = nt * 16 + (lane & 15);
        int rb  = (lane >> 4) * 8;                 // C/D: lanes 16-31 hold M = r+8
        for (int r = 0; r < 8; ++r)
          zbuf[(rb + r) * NPAD + col] = sigmoidf_(acc[r]);
      }
    }
    __syncthreads();

    // ---- phase 3: gate math, cell/hidden update, h -> Afrag + global ----
    for (int idx = tid; idx < 16 * H_; idx += blockDim.x) {
      int m = idx / H_, hc = idx % H_;
      float zi = zbuf[m * NPAD + hc];            // gate order i, f, g, o
      float zf = zbuf[m * NPAD + H_ + hc];
      float zg = zbuf[m * NPAD + 2 * H_ + hc];
      float zo = zbuf[m * NPAD + 3 * H_ + hc];
      float c  = zf * cbuf[idx] + zi * zg;
      cbuf[idx] = c;
      float h  = zo * sigmoidf_(c);
      store_afrag(Afrag, m, F_ + hc, f2bf(h));   // feed back for next step
      size_t o = (((size_t)inp * B_ + rowbase + m) * T_ + t) * (size_t)(2 * H_)
               + (size_t)(dir * H_) + hc;
      out[o] = h;                                // [2][B][T][2H] concat layout
    }
    __syncthreads();
  }
}

extern "C" void kernel_launch(void* const* d_in, const int* in_sizes, int n_in,
                              void* d_out, int out_size, void* d_ws, size_t ws_size,
                              hipStream_t stream) {
  (void)in_sizes; (void)n_in; (void)out_size; (void)d_ws; (void)ws_size;
  const float* context = (const float*)d_in[0];
  const float* question= (const float*)d_in[1];
  const float* Wf      = (const float*)d_in[2];
  const float* Rf      = (const float*)d_in[3];
  const float* bf      = (const float*)d_in[4];
  const float* Wb      = (const float*)d_in[5];
  const float* Rb      = (const float*)d_in[6];
  const float* bbias   = (const float*)d_in[7];
  float* out           = (float*)d_out;

  dim3 grid(64);    // 4 (input,dir) combos x 16 batch tiles of 16 rows
  dim3 block(256);  // 8 wave32 waves
  bilstm_scan_kernel<<<grid, block, SMEM_BYTES, stream>>>(
      context, question, Wf, Rf, bf, Wb, Rb, bbias, out);
}
